// DisparityNetworkStageN_22531398435048
// MI455X (gfx1250) — compile-verified, matrix operand order checked by reading, not run on hardware
//
#include <hip/hip_runtime.h>

typedef __attribute__((ext_vector_type(16))) _Float16 v16h;
typedef __attribute__((ext_vector_type(8)))  _Float16 v8h;
typedef __attribute__((ext_vector_type(4)))  _Float16 v4h;
typedef __attribute__((ext_vector_type(8)))  float    v8f;
typedef __attribute__((ext_vector_type(4)))  unsigned int u32x4;
typedef __attribute__((ext_vector_type(8)))  int      i32x8;
typedef __attribute__((ext_vector_type(4)))  int      i32x4;

#define NB   8
#define FH   128
#define FW   256
#define FC   16
#define ND   9
#define BH   512
#define BW   1024

__device__ __forceinline__ int iclamp(int v, int lo, int hi) {
    return v < lo ? lo : (v > hi ? hi : v);
}

// ---------------------------------------------------------------------------
// Kernel 1: antialiased linear downsample of residuals [B,512,1024] -> wflow
// [B,128,256], scaled by h/H = 0.25. Scale=4 triangle filter, 8 taps/dim,
// half-pixel centers (center = 4*i + 1.5, taps 4i-2 .. 4i+5), edge-clamped.
// ---------------------------------------------------------------------------
__global__ void wflow_kernel(const float* __restrict__ res, float* __restrict__ wflow) {
    int idx = blockIdx.x * 256 + threadIdx.x;            // over B*128*256
    if (idx >= NB * FH * FW) return;
    int x = idx & 255;
    int r = idx >> 8;
    int y = r & 127;
    int b = r >> 7;
    const float wt[8] = {0.125f*0.25f, 0.375f*0.25f, 0.625f*0.25f, 0.875f*0.25f,
                         0.875f*0.25f, 0.625f*0.25f, 0.375f*0.25f, 0.125f*0.25f};
    int by = 4 * y - 2, bx = 4 * x - 2;
    float acc = 0.f;
    for (int j = 0; j < 8; ++j) {
        int sy = iclamp(by + j, 0, BH - 1);
        const float* row = res + ((long)b * BH + sy) * BW;
        float accx = 0.f;
        for (int i = 0; i < 8; ++i) {
            int sx = iclamp(bx + i, 0, BW - 1);
            accx += wt[i] * row[sx];
        }
        acc += wt[j] * accx;
    }
    wflow[idx] = acc * 0.25f;   // * (h/H)
}

// ---------------------------------------------------------------------------
// Kernel 2: warp + L1 cost volume -> f16 [B,h,w,D]
// ---------------------------------------------------------------------------
__global__ void cost_kernel(const float* __restrict__ fl, const float* __restrict__ fr,
                            const float* __restrict__ wflow, _Float16* __restrict__ cost) {
    int idx = blockIdx.x * 256 + threadIdx.x;            // over B*h*w*D
    if (idx >= NB * FH * FW * ND) return;
    int d = idx % ND;
    int p = idx / ND;                                    // = (b*h + y)*w + x
    int x = p & 255;
    int r = p >> 8;
    int y = r & 127;
    int b = r >> 7;

    float shift = (float)(d - 4);
    float disp  = wflow[p] - shift;
    float xq    = (float)x - disp;
    float x0f   = floorf(xq);
    x0f = fminf(fmaxf(x0f, 0.f), (float)(FW - 2));
    float alpha = fminf(fmaxf(xq - x0f, 0.f), 1.f);
    int x0 = (int)x0f;

    const float4* l  = (const float4*)(fl + (long)p * FC);
    const float4* g0 = (const float4*)(fr + (((long)(b * FH + y)) * FW + x0) * FC);
    const float4* g1 = g0 + (FC / 4);                    // x0 + 1
    float c = 0.f;
    #pragma unroll
    for (int q = 0; q < FC / 4; ++q) {
        float4 a = l[q], u = g0[q], v = g1[q];
        c += fabsf(a.x - (u.x + alpha * (v.x - u.x)));
        c += fabsf(a.y - (u.y + alpha * (v.y - u.y)));
        c += fabsf(a.z - (u.z + alpha * (v.z - u.z)));
        c += fabsf(a.w - (u.w + alpha * (v.w - u.w)));
    }
    cost[idx] = (_Float16)c;
}

// ---------------------------------------------------------------------------
// Kernel 3: 3x3x3 conv over (h,w,D) as implicit GEMM on v_wmma_f32_16x16x32_f16.
//   M = voxels (16 per wave tile), K = 27*CI (zero-padded to mult of 32),
//   N = 16 (cols < CO live). Weights are bulk-copied global->LDS by the
//   Tensor Data Mover (one wave issues tensor_load_to_lds with a 1-row D#,
//   s_wait_tensorcnt 0, barrier), then converted f32->f16 into a [n][Kpad]
//   tile so each lane's B fragment is two contiguous 16B ds_load_b128 reads.
//
// A-fragment layout (16-bit A, wave32): lane row M = lane&15; lanes 0-15 hold
// halves K ka..ka+7 / ka+16..ka+23 with ka = kk, lanes 16-31 the same with
// ka = kk+8. For CI=4 each 4-half chunk is one tap x 4 channels -> a single
// 8-byte vector load at a precomputed neighbor offset (LDS table), zeroed
// branchlessly via a per-lane 27-bit validity mask. All loads are issued
// unconditionally (speculative but inside ws; pad offsets are 0) so no exec
// branching occurs and EXEC remains all-ones around every WMMA.
// C/D layout: vgpr r, lane L -> row r + 8*(L>=16), col L&15.
// ---------------------------------------------------------------------------
template<int CI, int CO, bool RELU, bool OUT_F16>
__global__ void conv3d_wmma_kernel(const _Float16* __restrict__ in,
                                   const float* __restrict__ wgt,
                                   const float* __restrict__ bias,
                                   _Float16* __restrict__ out16,
                                   float* __restrict__ out32) {
    constexpr int K  = 27 * CI;
    constexpr int KP = (K + 31) & ~31;
    __shared__ __align__(16) _Float16 Wlds[16 * KP];     // [n][k] f16 B tile
    __shared__ __align__(16) float    Wf32[K * CO];      // raw f32 weights (TDM dest)
    __shared__ __align__(16) int      offT[32];          // neighbor offsets per tap

    int tid = threadIdx.x;
    for (int i = tid; i < 16 * KP; i += 256) Wlds[i] = (_Float16)0.f;
    if (tid < 32) {
        int t = tid, off = 0;
        if (t < 27) {
            int td = t % 3, tt = t / 3, tx = tt % 3, ty = tt / 3;
            off = (((ty - 1) * FW + (tx - 1)) * ND + (td - 1)) * CI;
        }
        offT[t] = off;
    }

#if __has_builtin(__builtin_amdgcn_tensor_load_to_lds)
    if (tid < 32) {                                      // wave 0 only (uniform)
        constexpr unsigned NE = (unsigned)(K * CO);      // elements (f32)
        unsigned lds_addr = (unsigned)(size_t)(void*)&Wf32[0];
        unsigned long long ga = (unsigned long long)(size_t)wgt;
        u32x4 g0;
        g0[0] = 1u;                                      // count=1, user desc
        g0[1] = lds_addr;                                // lds_addr [63:32]
        g0[2] = (unsigned)(ga & 0xFFFFFFFFu);            // global_addr [95:64]
        g0[3] = (unsigned)((ga >> 32) & 0x1FFFFFFu)      // global_addr [120:96]
              | (2u << 30);                              // type=2 (image)
        i32x8 g1;
        g1[0] = (int)(2u << 16);                         // data_size=4B; no mask
        g1[1] = (int)((NE & 0xFFFFu) << 16);             // tensor_dim0 [63:48]
        g1[2] = (int)(((NE >> 16) & 0xFFFFu) | (1u << 16)); // dim0 hi | tensor_dim1=1
        g1[3] = (int)(NE << 16);                         // tile_dim0 [127:112]
        g1[4] = 1;                                       // tile_dim1=1, tile_dim2=0
        g1[5] = (int)NE;                                 // tensor_dim0_stride lo
        g1[6] = (int)((NE & 0xFFFFu) << 16);             // stride hi=0 | dim1_stride lo
        g1[7] = 0;
        i32x4 g2; g2[0] = 0; g2[1] = 0; g2[2] = 0; g2[3] = 0;
        i32x4 g3; g3[0] = 0; g3[1] = 0; g3[2] = 0; g3[3] = 0;
        i32x8 g4; g4[0] = 0; g4[1] = 0; g4[2] = 0; g4[3] = 0;
        g4[4] = 0; g4[5] = 0; g4[6] = 0; g4[7] = 0;
        __builtin_amdgcn_tensor_load_to_lds(g0, g1, g2, g3, g4, 0);
        __builtin_amdgcn_s_wait_tensorcnt(0);
    }
    __syncthreads();                                     // zeros + TDM data visible
    for (int i = tid; i < K * CO; i += 256) {
        int k  = i / CO;
        int co = i - k * CO;
        Wlds[co * KP + k] = (_Float16)Wf32[i];           // wgt flat: (t*CI+ci)*CO+co
    }
#else
    __syncthreads();
    for (int i = tid; i < K * CO; i += 256) {
        int k  = i / CO;
        int co = i - k * CO;
        Wlds[co * KP + k] = (_Float16)wgt[i];
    }
#endif
    __syncthreads();

    int wave = tid >> 5;
    int lane = tid & 31;
    int hi   = lane >> 4;
    int n    = lane & 15;

    int m_base = (blockIdx.x * 8 + wave) * 16;           // grid divides V exactly
    int m = m_base + n;                                  // A row for this lane
    int d = m % ND;
    int r = m / ND;
    int x = r & 255; r >>= 8;
    int y = r & 127;
    int b = r >> 7;
    const _Float16* pbase = in + (long)m * CI;

    // per-lane tap validity mask (27 bits); bits 27..31 stay 0
    int vmask = 0;
    #pragma unroll
    for (int t = 0; t < 27; ++t) {
        int td = t % 3, tt = t / 3, tx = tt % 3, ty = tt / 3;
        bool ok = ((unsigned)(y + ty - 1) < (unsigned)FH) &&
                  ((unsigned)(x + tx - 1) < (unsigned)FW) &&
                  ((unsigned)(d + td - 1) < (unsigned)ND);
        if (ok) vmask |= (1 << t);
    }

    union AV { v16h v; v4h q[4]; _Float16 e[16]; };
    union BV { v16h v; v8h h2[2]; };
    const int cofs[4] = {0, 4, 16, 20};                  // half-offset of each 4-chunk

    v8f c = {};
    #pragma unroll
    for (int kk = 0; kk < KP; kk += 32) {
        int ka = kk + hi * 8;
        AV a;
        if constexpr (CI == 4) {
            #pragma unroll
            for (int g = 0; g < 4; ++g) {
                int t   = (ka + cofs[g]) >> 2;           // one tap per chunk
                int off = offT[t & 31];
                v4h v   = *(const v4h*)(pbase + off);    // unconditional 8B load
                bool ok = (vmask >> (t & 31)) & 1;
                v4h z = {};
                a.q[g] = ok ? v : z;                     // cndmask, no branch
            }
        } else {                                         // CI == 1: 4 taps per chunk
            #pragma unroll
            for (int g = 0; g < 4; ++g) {
                int ks = ka + cofs[g];                   // multiple of 4
                int4 o = *(const int4*)&offT[ks];        // 4 offsets, one b128
                _Float16 v0 = pbase[o.x];                // unconditional loads
                _Float16 v1 = pbase[o.y];
                _Float16 v2 = pbase[o.z];
                _Float16 v3 = pbase[o.w];
                _Float16 z  = (_Float16)0.f;
                a.e[g * 4 + 0] = ((vmask >> ((ks + 0) & 31)) & 1) ? v0 : z;
                a.e[g * 4 + 1] = ((vmask >> ((ks + 1) & 31)) & 1) ? v1 : z;
                a.e[g * 4 + 2] = ((vmask >> ((ks + 2) & 31)) & 1) ? v2 : z;
                a.e[g * 4 + 3] = ((vmask >> ((ks + 3) & 31)) & 1) ? v3 : z;
            }
        }
        BV bf;
        const _Float16* wrow = &Wlds[n * KP];
        bf.h2[0] = *(const v8h*)(wrow + ka);             // ds_load_b128
        bf.h2[1] = *(const v8h*)(wrow + ka + 16);        // ds_load_b128
        c = __builtin_amdgcn_wmma_f32_16x16x32_f16(false, a.v, false, bf.v,
                                                   (short)0, c, false, false);
    }

    float bn = (n < CO) ? bias[n] : 0.f;
    #pragma unroll
    for (int rr = 0; rr < 8; ++rr) {
        long mm = (long)m_base + hi * 8 + rr;
        float v = c[rr] + bn;
        if (RELU) v = fmaxf(v, 0.f);
        if (n < CO) {
            if (OUT_F16) out16[mm * CO + n] = (_Float16)v;
            else         out32[mm * CO + n] = v;
        }
    }
}

// ---------------------------------------------------------------------------
// Kernel 4: softmax(-x) over D, disparity regression -> pred [B,h,w]
// ---------------------------------------------------------------------------
__global__ void softreg_kernel(const float* __restrict__ fin, float* __restrict__ pred) {
    int p = blockIdx.x * 256 + threadIdx.x;
    if (p >= NB * FH * FW) return;
    const float* xr = fin + (long)p * ND;
    float xs[ND];
    float mx = -3.4e38f;
    #pragma unroll
    for (int d = 0; d < ND; ++d) {
        xs[d] = -xr[d];
        mx = fmaxf(mx, xs[d]);
    }
    float s = 0.f, acc = 0.f;
    #pragma unroll
    for (int d = 0; d < ND; ++d) {
        float e = __expf(xs[d] - mx);
        s   += e;
        acc += e * (float)(d - 4);
    }
    pred[p] = acc / s;
}

// ---------------------------------------------------------------------------
// Kernel 5: bilinear upsample pred (128,256)->(512,1024), add residuals.
// ---------------------------------------------------------------------------
__global__ void upsample_add_kernel(const float* __restrict__ pred,
                                    const float* __restrict__ res,
                                    float* __restrict__ out) {
    int idx = blockIdx.x * 256 + threadIdx.x;
    if (idx >= NB * BH * BW) return;
    int X = idx & 1023;
    int r = idx >> 10;
    int Y = r & 511;
    int b = r >> 9;
    float sy = 0.25f * (float)Y - 0.375f;
    float sx = 0.25f * (float)X - 0.375f;
    float y0f = floorf(sy), x0f = floorf(sx);
    float ay = sy - y0f, ax = sx - x0f;
    int y0 = iclamp((int)y0f,     0, FH - 1);
    int y1 = iclamp((int)y0f + 1, 0, FH - 1);
    int x0 = iclamp((int)x0f,     0, FW - 1);
    int x1 = iclamp((int)x0f + 1, 0, FW - 1);
    const float* pb = pred + (long)b * FH * FW;
    float v00 = pb[y0 * FW + x0], v01 = pb[y0 * FW + x1];
    float v10 = pb[y1 * FW + x0], v11 = pb[y1 * FW + x1];
    float v0 = v00 + ax * (v01 - v00);
    float v1 = v10 + ax * (v11 - v10);
    out[idx] = v0 + ay * (v1 - v0) + res[idx];
}

// ---------------------------------------------------------------------------
extern "C" void kernel_launch(void* const* d_in, const int* in_sizes, int n_in,
                              void* d_out, int out_size, void* d_ws, size_t ws_size,
                              hipStream_t stream) {
    const float* residuals = (const float*)d_in[0];
    const float* feats_l   = (const float*)d_in[1];
    const float* feats_r   = (const float*)d_in[2];
    const float* w0 = (const float*)d_in[3];  const float* b0 = (const float*)d_in[4];
    const float* w1 = (const float*)d_in[5];  const float* b1 = (const float*)d_in[6];
    const float* w2 = (const float*)d_in[7];  const float* b2 = (const float*)d_in[8];
    const float* w3 = (const float*)d_in[9];  const float* b3 = (const float*)d_in[10];
    const float* wf = (const float*)d_in[11]; const float* bf = (const float*)d_in[12];

    char* ws = (char*)d_ws;
    // Workspace layout (51 MB total; conv halo overreads stay inside ws):
    float*    wflow = (float*)   (ws + 0);                 //  1.0 MB [B,h,w] f32
    _Float16* cost  = (_Float16*)(ws + (1u  << 20));       //  4.5 MB [B,h,w,D] f16
    _Float16* actA  = (_Float16*)(ws + (6u  << 20));       // 18.0 MB [B,h,w,D,4] f16
    _Float16* actB  = (_Float16*)(ws + (24u << 20));       // 18.0 MB
    float*    fin   = (float*)   (ws + (42u << 20));       //  9.0 MB [B,h,w,D] f32
    float*    pred  = wflow;                               // reuse (wflow dead)

    const int nPix   = NB * FH * FW;                        // 262144
    const int nVox   = nPix * ND;                           // 2359296
    const int nOut   = NB * BH * BW;                        // 4194304
    const int convBl = nVox / (8 * 16);                     // 18432 (exact)

    wflow_kernel<<<nPix / 256, 256, 0, stream>>>(residuals, wflow);
    cost_kernel <<<nVox / 256, 256, 0, stream>>>(feats_l, feats_r, wflow, cost);

    conv3d_wmma_kernel<1, 4, true,  true ><<<convBl, 256, 0, stream>>>(cost, w0, b0, actA, nullptr);
    conv3d_wmma_kernel<4, 4, true,  true ><<<convBl, 256, 0, stream>>>(actA, w1, b1, actB, nullptr);
    conv3d_wmma_kernel<4, 4, true,  true ><<<convBl, 256, 0, stream>>>(actB, w2, b2, actA, nullptr);
    conv3d_wmma_kernel<4, 4, true,  true ><<<convBl, 256, 0, stream>>>(actA, w3, b3, actB, nullptr);
    conv3d_wmma_kernel<4, 1, false, false><<<convBl, 256, 0, stream>>>(actB, wf, bf, nullptr, fin);

    softreg_kernel     <<<nPix / 256, 256, 0, stream>>>(fin, pred);
    upsample_add_kernel<<<nOut / 256, 256, 0, stream>>>(pred, residuals, (float*)d_out);
}
